// AttentionContextEncoder_16492674416952
// MI455X (gfx1250) — compile-verified, matrix operand order checked by reading, not run on hardware
//
#include <hip/hip_runtime.h>
#include <hip/hip_bf16.h>

// =====================  CDNA5 fused AttentionContextEncoder  =====================
// M=16 batch rows per wave32; every dense layer runs as v_wmma_f32_16x16x32_f16
// with A-fragments from LDS (row-major activations) and B-fragments streamed from
// global d_ws (weights pre-transposed/padded to f16 [N][Kpad] by a prep kernel;
// ~308KB total, stays hot in the 192MB L2; warmed with global_prefetch_b8).

typedef _Float16 v16h __attribute__((ext_vector_type(16)));
typedef float    v8f  __attribute__((ext_vector_type(8)));

#define TILE  16
#define WAVES 2

// half-element offsets inside d_ws for the transposed/padded f16 weights
enum : int {
  OFF_WVP = 0,       // Kpad=32 N=32   (14x32)
  OFF_WAP = 1024,    // Kpad=32 N=64   (17x64)
  OFF_WPP = 3072,    // Kpad=64 N=32   (51x32)
  OFF_WSP = 5120,    // Kpad=32 N=16   (7x16)
  OFF_WTP = 5632,    // Kpad=32 N=16   (10x16)
  OFF_WVU = 6144,    // Kpad=32 N=64
  OFF_WAU = 8192,    // Kpad=64 N=64
  OFF_WPU = 12288,   // Kpad=32 N=64
  OFF_WSU = 14336,   // Kpad=32 N=64  (16->32 pad)
  OFF_WTU = 16384,   // Kpad=32 N=64  (16->32 pad)
  OFF_WQ  = 18432,   // 64x64
  OFF_WK  = 22528,
  OFF_WV  = 26624,
  OFF_WO  = 30720,
  OFF_W1  = 34816,   // Kpad=320 N=256
  OFF_W2  = 116736,  // Kpad=256 N=160
  WS_HALFS = 157696
};

// -------- prep: transpose f32 weight (K x N, row-major) -> f16 [N][Kpad], zero-padded K
__global__ void prep_transpose(const float* __restrict__ src, _Float16* __restrict__ dst,
                               int K, int N, int Kpad) {
  int idx = blockIdx.x * blockDim.x + threadIdx.x;
  if (idx >= N * Kpad) return;
  int n = idx / Kpad, k = idx % Kpad;
  dst[idx] = (k < K) ? (_Float16)src[k * N + n] : (_Float16)0.f;
}

// -------- WMMA helpers (contiguous k = 16*(lane/16) + s, same in A and B) --------
__device__ __forceinline__ v8f wmma32(v16h a, v16h b, v8f c) {
  return __builtin_amdgcn_wmma_f32_16x16x32_f16(false, a, false, b, (short)0, c, false, false);
}
// A 16x32 f16: lane m=lane&15 holds K-halves [kb..kb+15], lane-group selects K half
__device__ __forceinline__ v16h frag_a(const _Float16* base, int strideH, int kOff, int lane) {
  int m  = lane & 15;
  int kb = kOff + ((lane >> 4) << 4);
  return *(const v16h*)(base + m * strideH + kb);
}
// B 32x16 f16 from transposed weight [N][Kpad]: lane n=lane&15 holds its column's K-halves
__device__ __forceinline__ v16h frag_b(const _Float16* wt, int Kpad, int nBase, int kOff, int lane) {
  int n  = nBase + (lane & 15);
  int kb = kOff + ((lane >> 4) << 4);
  return *(const v16h*)(wt + n * Kpad + kb);
}

// compile-time specialized tile-gemm: D = act(A @ Wt + bias) scattered to LDS f16 and/or f32
template <int NT, int KS, bool RELU, bool HAS16, bool HAS32>
__device__ __forceinline__ void gemm(const _Float16* A, int strideA,
                                     const _Float16* WT, int Kpad,
                                     const float* bias,
                                     _Float16* O16, int s16,
                                     float* O32, int s32, int lane) {
  const int n  = lane & 15;
  const int mh = (lane >> 4) << 3;
#pragma unroll
  for (int nt = 0; nt < NT; ++nt) {
    v8f acc = {};
#pragma unroll
    for (int ks = 0; ks < KS; ++ks) {
      v16h a = frag_a(A, strideA, ks * 32, lane);
      v16h b = frag_b(WT, Kpad, nt * 16, ks * 32, lane);
      acc = wmma32(a, b, acc);
    }
    float bb = bias[nt * 16 + n];
#pragma unroll
    for (int v = 0; v < 8; ++v) {
      float val = acc[v] + bb;
      if (RELU) val = fmaxf(val, 0.f);
      int r = mh + v, c = nt * 16 + n;
      if (HAS16) O16[r * s16 + c] = (_Float16)val;
      if (HAS32) O32[r * s32 + c] = val;
    }
  }
}

__device__ __forceinline__ void load_input(const float* __restrict__ in, int inDim, int r0,
                                           _Float16* xin, int lane) {
  for (int idx = lane; idx < 16 * 64; idx += 32) {
    int row = idx >> 6, k = idx & 63;
    float v = (k < inDim) ? in[(r0 + row) * inDim + k] : 0.f;
    xin[idx] = (_Float16)v;
  }
}

__device__ __forceinline__ void zero_pad_cols(_Float16* work, int lane) {
  // zero cols 16..31 of the 16x(stride 320) work tile (for Ppad=32 layers with P=16)
  for (int idx = lane; idx < 256; idx += 32) {
    int row = idx >> 4, c = 16 + (idx & 15);
    work[row * 320 + c] = (_Float16)0.f;
  }
}

struct KParams {
  const float *visual, *audio, *pose, *spatial, *timex;
  const float *bvp, *bap, *bpp, *bsp, *btp;
  const float *bvu, *bau, *bpu, *bsu, *btu;
  const float *bq, *bk, *bv, *bo, *gamma, *beta, *b1, *b2;
  const _Float16* W;
  float* out;
  int nrows;
};

__global__ __launch_bounds__(WAVES * 32, 1)
void ace_fused_kernel(KParams p) {
  extern __shared__ char smem_raw[];
  const int lane = threadIdx.x & 31;
  const int wave = threadIdx.x >> 5;
  // per-wave private 80KB LDS region
  _Float16* sbase = (_Float16*)smem_raw + wave * 40960;
  _Float16* work  = sbase;          // [16][320] f16 : proj tmp / attn-o / normalized flat h
  _Float16* xa    = work + 5120;    // [5][16][64] f16 : x tokens (gemm A source)
  _Float16* qkv   = xa + 5120;      // [3][5][16][64] f16
  _Float16* fused = qkv + 15360;    // [16][256] f16
  _Float16* xin   = fused + 4096;   // [16][64] f16 : padded raw input tile
  float*    xf    = (float*)(xin + 1024); // [5][16][64] f32 : x residual, then h

  int tile = blockIdx.x * WAVES + wave;
  int r0 = tile * TILE;
  if (r0 >= p.nrows) return;

  // warm L2/WGP$ for the big MLP weight images (gfx1250 global_prefetch_b8)
  __builtin_prefetch(p.W + OFF_W1 + lane * 2560, 0, 1);
  __builtin_prefetch(p.W + OFF_W2 + lane * 1280, 0, 1);

  // ---------------- modality branches: relu(in@Wp+b) @ Wu + b -> x tokens -------------
  load_input(p.visual, 14, r0, xin, lane);
  gemm<2, 1, true,  true, false>(xin, 64, p.W + OFF_WVP, 32, p.bvp, work, 320, (float*)0, 0, lane);
  gemm<4, 1, false, true, true >(work, 320, p.W + OFF_WVU, 32, p.bvu, xa + 0 * 1024, 64, xf + 0 * 1024, 64, lane);

  load_input(p.audio, 17, r0, xin, lane);
  gemm<4, 1, true,  true, false>(xin, 64, p.W + OFF_WAP, 32, p.bap, work, 320, (float*)0, 0, lane);
  gemm<4, 2, false, true, true >(work, 320, p.W + OFF_WAU, 64, p.bau, xa + 1 * 1024, 64, xf + 1 * 1024, 64, lane);

  load_input(p.pose, 51, r0, xin, lane);
  gemm<2, 2, true,  true, false>(xin, 64, p.W + OFF_WPP, 64, p.bpp, work, 320, (float*)0, 0, lane);
  gemm<4, 1, false, true, true >(work, 320, p.W + OFF_WPU, 32, p.bpu, xa + 2 * 1024, 64, xf + 2 * 1024, 64, lane);

  load_input(p.spatial, 7, r0, xin, lane);
  zero_pad_cols(work, lane);
  gemm<1, 1, true,  true, false>(xin, 64, p.W + OFF_WSP, 32, p.bsp, work, 320, (float*)0, 0, lane);
  gemm<4, 1, false, true, true >(work, 320, p.W + OFF_WSU, 32, p.bsu, xa + 3 * 1024, 64, xf + 3 * 1024, 64, lane);

  load_input(p.timex, 10, r0, xin, lane);
  zero_pad_cols(work, lane);
  gemm<1, 1, true,  true, false>(xin, 64, p.W + OFF_WTP, 32, p.btp, work, 320, (float*)0, 0, lane);
  gemm<4, 1, false, true, true >(work, 320, p.W + OFF_WTU, 32, p.btu, xa + 4 * 1024, 64, xf + 4 * 1024, 64, lane);

  // ---------------- QKV projections (E=64 -> H*D=64), per token --------------------
  for (int tok = 0; tok < 5; ++tok) {
    gemm<4, 2, false, true, false>(xa + tok * 1024, 64, p.W + OFF_WQ, 64, p.bq, qkv + (0 * 5 + tok) * 1024, 64, (float*)0, 0, lane);
    gemm<4, 2, false, true, false>(xa + tok * 1024, 64, p.W + OFF_WK, 64, p.bk, qkv + (1 * 5 + tok) * 1024, 64, (float*)0, 0, lane);
    gemm<4, 2, false, true, false>(xa + tok * 1024, 64, p.W + OFF_WV, 64, p.bv, qkv + (2 * 5 + tok) * 1024, 64, (float*)0, 0, lane);
  }

  // ------- tiny 5x5 softmax attention: VALU, vectorized q/k/v access (b128 DS loads) --
  for (int u = lane; u < 64; u += 32) {
    int row = u & 15, h = u >> 4;
    float sc[5][5];
#pragma unroll
    for (int qi = 0; qi < 5; ++qi) {
      v16h qv = *(const v16h*)(qkv + ((0 * 5 + qi) * 16 + row) * 64 + h * 16);
#pragma unroll
      for (int ki = 0; ki < 5; ++ki) {
        v16h kv = *(const v16h*)(qkv + ((1 * 5 + ki) * 16 + row) * 64 + h * 16);
        float d = 0.f;
#pragma unroll
        for (int e = 0; e < 16; ++e) d += (float)qv[e] * (float)kv[e];
        sc[qi][ki] = d * 0.25f; // 1/sqrt(16)
      }
    }
#pragma unroll
    for (int qi = 0; qi < 5; ++qi) {
      float mx = sc[qi][0];
#pragma unroll
      for (int ki = 1; ki < 5; ++ki) mx = fmaxf(mx, sc[qi][ki]);
      float s = 0.f;
#pragma unroll
      for (int ki = 0; ki < 5; ++ki) { float e = __expf(sc[qi][ki] - mx); sc[qi][ki] = e; s += e; }
      float inv = 1.f / s;
      float o[16];
#pragma unroll
      for (int e = 0; e < 16; ++e) o[e] = 0.f;
#pragma unroll
      for (int ki = 0; ki < 5; ++ki) {
        v16h vv = *(const v16h*)(qkv + ((2 * 5 + ki) * 16 + row) * 64 + h * 16);
        float w = sc[qi][ki];
#pragma unroll
        for (int e = 0; e < 16; ++e) o[e] += w * (float)vv[e];
      }
      v16h ov;
#pragma unroll
      for (int e = 0; e < 16; ++e) ov[e] = (_Float16)(o[e] * inv);
      *(v16h*)(work + row * 320 + qi * 64 + h * 16) = ov;  // 32B vector store
    }
  }

  // ---------------- output projection + residual: h = x + o@Wo + bo (f32 in xf) ------
  {
    int n = lane & 15, mh = (lane >> 4) << 3;
#pragma unroll
    for (int tok = 0; tok < 5; ++tok) {
#pragma unroll
      for (int nt = 0; nt < 4; ++nt) {
        v8f acc = {};
#pragma unroll
        for (int ks = 0; ks < 2; ++ks) {
          v16h a = frag_a(work + tok * 64, 320, ks * 32, lane);
          v16h b = frag_b(p.W + OFF_WO, 64, nt * 16, ks * 32, lane);
          acc = wmma32(a, b, acc);
        }
        float bb = p.bo[nt * 16 + n];
#pragma unroll
        for (int v = 0; v < 8; ++v) {
          int r = mh + v, c = nt * 16 + n;
          float* hp = xf + (tok * 16 + r) * 64 + c;
          *hp = *hp + acc[v] + bb;
        }
      }
    }
  }

  // ------- LayerNorm(eps=1e-3) over E, vectorized write of normalized f16 flat -------
  for (int u = lane; u < 80; u += 32) {
    int row = u & 15, tok = u >> 4;
    const float* hr = xf + (tok * 16 + row) * 64;
    float mu = 0.f;
#pragma unroll
    for (int e = 0; e < 64; ++e) mu += hr[e];
    mu *= (1.f / 64.f);
    float var = 0.f;
#pragma unroll
    for (int e = 0; e < 64; ++e) { float d = hr[e] - mu; var += d * d; }
    var *= (1.f / 64.f);
    float rinv = rsqrtf(var + 1e-3f);
#pragma unroll
    for (int c = 0; c < 4; ++c) {
      v16h ov;
#pragma unroll
      for (int e2 = 0; e2 < 16; ++e2) {
        int e = c * 16 + e2;
        ov[e2] = (_Float16)((hr[e] - mu) * rinv * p.gamma[e] + p.beta[e]);
      }
      *(v16h*)(work + row * 320 + tok * 64 + c * 16) = ov;
    }
  }

  // ---------------- fusion MLP: 320 -> 256 (relu) -> 160 (relu) ----------------------
  gemm<16, 10, true, true, false>(work, 320, p.W + OFF_W1, 320, p.b1, fused, 256, (float*)0, 0, lane);

  {
    int n = lane & 15, mh = (lane >> 4) << 3;
#pragma unroll
    for (int nt = 0; nt < 10; ++nt) {
      v8f acc = {};
#pragma unroll
      for (int ks = 0; ks < 8; ++ks) {
        v16h a = frag_a(fused, 256, ks * 32, lane);
        v16h b = frag_b(p.W + OFF_W2, 256, nt * 16, ks * 32, lane);
        acc = wmma32(a, b, acc);
      }
      float bb = p.b2[nt * 16 + n];
#pragma unroll
      for (int v = 0; v < 8; ++v) {
        float val = fmaxf(acc[v] + bb, 0.f);
        p.out[(r0 + mh + v) * 160 + nt * 16 + n] = val;
      }
    }
  }
}

extern "C" void kernel_launch(void* const* d_in, const int* in_sizes, int n_in,
                              void* d_out, int out_size, void* d_ws, size_t ws_size,
                              hipStream_t stream) {
  (void)n_in; (void)out_size;
  if (ws_size < (size_t)WS_HALFS * sizeof(_Float16)) return;
  _Float16* W = (_Float16*)d_ws;

  auto launchT = [&](const void* src, int off, int K, int N, int Kpad) {
    int total = N * Kpad;
    prep_transpose<<<(total + 255) / 256, 256, 0, stream>>>((const float*)src, W + off, K, N, Kpad);
  };
  // transposed/padded f16 weight staging (stays hot in L2 across the main kernel)
  launchT(d_in[5],  OFF_WVP, 14, 32, 32);
  launchT(d_in[7],  OFF_WAP, 17, 64, 32);
  launchT(d_in[9],  OFF_WPP, 51, 32, 64);
  launchT(d_in[11], OFF_WSP, 7, 16, 32);
  launchT(d_in[13], OFF_WTP, 10, 16, 32);
  launchT(d_in[15], OFF_WVU, 32, 64, 32);
  launchT(d_in[17], OFF_WAU, 64, 64, 64);
  launchT(d_in[19], OFF_WPU, 32, 64, 32);
  launchT(d_in[21], OFF_WSU, 16, 64, 32);
  launchT(d_in[23], OFF_WTU, 16, 64, 32);
  launchT(d_in[25], OFF_WQ, 64, 64, 64);
  launchT(d_in[27], OFF_WK, 64, 64, 64);
  launchT(d_in[29], OFF_WV, 64, 64, 64);
  launchT(d_in[31], OFF_WO, 64, 64, 64);
  launchT(d_in[35], OFF_W1, 320, 256, 320);
  launchT(d_in[37], OFF_W2, 256, 160, 256);

  KParams p;
  p.visual = (const float*)d_in[0];
  p.audio  = (const float*)d_in[1];
  p.pose   = (const float*)d_in[2];
  p.spatial= (const float*)d_in[3];
  p.timex  = (const float*)d_in[4];
  p.bvp = (const float*)d_in[6];  p.bap = (const float*)d_in[8];
  p.bpp = (const float*)d_in[10]; p.bsp = (const float*)d_in[12]; p.btp = (const float*)d_in[14];
  p.bvu = (const float*)d_in[16]; p.bau = (const float*)d_in[18];
  p.bpu = (const float*)d_in[20]; p.bsu = (const float*)d_in[22]; p.btu = (const float*)d_in[24];
  p.bq = (const float*)d_in[26];  p.bk = (const float*)d_in[28];
  p.bv = (const float*)d_in[30];  p.bo = (const float*)d_in[32];
  p.gamma = (const float*)d_in[33]; p.beta = (const float*)d_in[34];
  p.b1 = (const float*)d_in[36];  p.b2 = (const float*)d_in[38];
  p.W = W;
  p.out = (float*)d_out;
  p.nrows = in_sizes[0] / 14;

  int tiles  = (p.nrows + TILE - 1) / TILE;
  int blocks = (tiles + WAVES - 1) / WAVES;
  size_t shmem = (size_t)WAVES * 81920; // 80KB per wave, 160KB of the WGP's 320KB
  ace_fused_kernel<<<blocks, WAVES * 32, shmem, stream>>>(p);
}
